// ScaledDotProductAttention_1898375545046
// MI455X (gfx1250) — matrix operand order repaired and checked by reading, not compiled, not run on hardware
//
#include <hip/hip_runtime.h>

// CDNA5 / gfx1250 attention: S=QK^T (WMMA f16->f32), masked softmax, O=PV (WMMA),
// full attn matrix materialized (memory-bound on the 536MB attn write).

typedef __attribute__((ext_vector_type(16))) _Float16 v16h;
typedef __attribute__((ext_vector_type(8)))  float    v8f;

#define B_ 32
#define L_ 2048
#define D_ 128
#define INV_T 0.08838834764831845f   // 1 / sqrt(128)
#define MASK_FILL -1e32f
#define SROW 2052                    // padded LDS row stride (2052 % 64 == 4)

__global__ __launch_bounds__(256)
void sdpa_wmma_kernel(const float* __restrict__ q,
                      const float* __restrict__ k,
                      const float* __restrict__ v,
                      float* __restrict__ out,
                      float* __restrict__ attn)
{
    extern __shared__ float lds[];
    float* S       = lds;                 // 16 * SROW scores -> probabilities
    float* rowmax  = lds + 16 * SROW;     // 16
    float* rowinv  = rowmax + 16;         // 16 (row sum, then its inverse)
    float* partmax = rowinv + 16;         // 8 waves * 16 rows

    const int tid  = threadIdx.x;
    const int lane = tid & 31;
    const int wave = tid >> 5;
    const int t15  = lane & 15;
    const int half = lane >> 4;           // 0: lanes 0-15, 1: lanes 16-31

    const int b  = blockIdx.x >> 7;       // / (L/16)
    const int m0 = (blockIdx.x & 127) << 4;

    const float* qb = q + ((size_t)b * L_) * D_;
    const float* kb = k + ((size_t)b * L_) * D_;
    const float* vb = v + ((size_t)b * L_) * D_;

    // ---------------- Phase 1: S = mask(Q K^T * invT), track row max ----------------
    // A fragment (Q rows m0..m0+15), lane = M row, half selects 16-wide K block (k = c*32 + half*16 + s).
    v16h aQ[4];
    {
        const float* qrow = qb + (size_t)(m0 + t15) * D_;
        #pragma unroll
        for (int c = 0; c < 4; ++c) {
            const int k0 = c * 32 + half * 16;
            #pragma unroll
            for (int j = 0; j < 8; ++j) {
                aQ[c][j]     = (_Float16)qrow[k0 + j];        // K = c*32 + half*16 + j
                aQ[c][j + 8] = (_Float16)qrow[k0 + 8 + j];    // K = c*32 + half*16 + 8 + j
            }
        }
    }

    float rm[8];
    #pragma unroll
    for (int r = 0; r < 8; ++r) rm[r] = -3.4e38f;

    for (int it = 0; it < 16; ++it) {
        const int col_base = ((it << 3) + wave) << 4;   // this wave's 16-col tile
        const int n = col_base + t15;                   // lane's key index (B-matrix N)
        const float* krow = kb + (size_t)n * D_ + half * 16;
        v8f acc = {};
        #pragma unroll
        for (int c = 0; c < 4; ++c) {
            v16h bK;  // B layout: lane = N, element j = K[n][c*32 + half*16 + j]
            #pragma unroll
            for (int j = 0; j < 16; ++j) bK[j] = (_Float16)krow[c * 32 + j];
            acc = __builtin_amdgcn_wmma_f32_16x16x32_f16(false, aQ[c], false, bK,
                                                         (short)0, acc, false, false);
        }
        // C/D layout: VGPR r -> row (r + 8*half), col = t15 within tile
        #pragma unroll
        for (int r = 0; r < 8; ++r) {
            const int ml = r + half * 8;
            float s = acc[r] * INV_T;
            if (m0 + ml == n) s = MASK_FILL;            // analytic ~eye mask
            S[ml * SROW + n] = s;
            rm[r] = fmaxf(rm[r], s);
        }
    }
    // butterfly max over the 16-lane half (all N columns this wave saw)
    #pragma unroll
    for (int m = 1; m < 16; m <<= 1) {
        #pragma unroll
        for (int r = 0; r < 8; ++r)
            rm[r] = fmaxf(rm[r], __shfl_xor(rm[r], m, 32));
    }
    if (t15 == 0) {
        #pragma unroll
        for (int r = 0; r < 8; ++r)
            partmax[wave * 16 + half * 8 + r] = rm[r];
    }
    __syncthreads();
    if (tid < 16) {
        float mx = partmax[tid];
        #pragma unroll
        for (int w = 1; w < 8; ++w) mx = fmaxf(mx, partmax[w * 16 + tid]);
        rowmax[tid] = mx;
    }
    __syncthreads();

    // ---------------- Phase 2: P = exp(S - rowmax), row sums ----------------
    {
        const int row = tid >> 4;          // 16 threads per row (same half-wave)
        const int g   = tid & 15;
        const float mx = rowmax[row];
        float* srow = S + row * SROW;
        float sum = 0.f;
        #pragma unroll 4
        for (int i = 0; i < 128; ++i) {
            const int col = g + (i << 4);
            const float p = __expf(srow[col] - mx);
            srow[col] = p;
            sum += p;
        }
        #pragma unroll
        for (int m = 1; m < 16; m <<= 1) sum += __shfl_xor(sum, m, 32);
        if (g == 0) rowinv[row] = 1.0f / sum;
    }
    __syncthreads();

    // ---------------- Phase 3a: stream normalized attn to HBM (the memory floor) ----
    {
        float* arow0 = attn + ((size_t)b * L_ + m0) * (size_t)L_;
        for (int r = 0; r < 16; ++r) {
            const float inv = rowinv[r];
            const float* srow = S + r * SROW;
            float* dst = arow0 + (size_t)r * L_;
            for (int c0 = tid * 4; c0 < L_; c0 += 1024) {
                float4 pv;
                pv.x = srow[c0 + 0] * inv;
                pv.y = srow[c0 + 1] * inv;
                pv.z = srow[c0 + 2] * inv;
                pv.w = srow[c0 + 3] * inv;
                *(float4*)(dst + c0) = pv;
            }
        }
    }

    // ---------------- Phase 3b: O = (P @ V) * rowinv, one 16-wide D slice per wave ----
    {
        const int dcol = (wave << 4) + t15;
        v8f o = {};
        for (int kc = 0; kc < 64; ++kc) {
            v16h aP;   // A layout from LDS (unnormalized P)
            const float* prow = S + t15 * SROW + kc * 32 + half * 16;
            #pragma unroll
            for (int j = 0; j < 8; ++j) {
                aP[j]     = (_Float16)prow[j];
                aP[j + 8] = (_Float16)prow[8 + j];
            }
            v16h bV;   // B layout: element j = V[kc*32 + half*16 + j][dcol] (lane-coalesced)
            const float* vk = vb + (size_t)(kc * 32 + half * 16) * D_ + dcol;
            #pragma unroll
            for (int j = 0; j < 16; ++j) bV[j] = (_Float16)vk[(size_t)j * D_];
            o = __builtin_amdgcn_wmma_f32_16x16x32_f16(false, aP, false, bV,
                                                       (short)0, o, false, false);
        }
        #pragma unroll
        for (int r = 0; r < 8; ++r) {
            const int ml = r + half * 8;
            out[((size_t)b * L_ + (m0 + ml)) * D_ + dcol] = o[r] * rowinv[ml];
        }
    }
}

extern "C" void kernel_launch(void* const* d_in, const int* in_sizes, int n_in,
                              void* d_out, int out_size, void* d_ws, size_t ws_size,
                              hipStream_t stream) {
    (void)in_sizes; (void)n_in; (void)d_ws; (void)ws_size; (void)out_size;
    const float* q = (const float*)d_in[0];
    const float* k = (const float*)d_in[1];
    const float* v = (const float*)d_in[2];
    // d_in[3] is diag_mask (~eye, broadcast) — applied analytically in-kernel.

    float* out  = (float*)d_out;                      // [B, L, D]
    float* attn = out + (size_t)B_ * L_ * D_;         // [B, L, L] follows in tuple order

    const size_t shmem = (size_t)(16 * SROW + 16 + 16 + 128) * sizeof(float); // ~132 KB (<320 KB WGP LDS)
    dim3 grid(B_ * (L_ / 16));                        // 4096 workgroups
    dim3 block(256);                                  // 8 wave32 waves
    sdpa_wmma_kernel<<<grid, block, shmem, stream>>>(q, k, v, out, attn);
}
